// TubeACARHead_858993459760
// MI455X (gfx1250) — matrix operand order repaired and checked by reading, not compiled, not run on hardware
//
#include <hip/hip_runtime.h>

// ---------------------------------------------------------------------------
// TubeACARHead for MI455X (gfx1250, wave32).
// All dense stages run as implicit GEMMs on V_WMMA_F32_16X16X32_BF16 with
// LDS-staged, pre-converted bf16 tiles (f32 -> bf16 RNE once per element,
// f32 accumulation). Per-column gather state (div/mod by runtime dims) is
// hoisted out of the K loop; inner-loop B gathers are branchless
// clamp+select. Each wave owns two 16x16 N-tiles (block tile 32x128) so the
// K loop issues 2 WMMAs per barrier pair.
// Irregular stages (ROI-align, 10x10 attention softmax, norms, pools) are
// VALU kernels with wave32 shuffle reductions.
// ---------------------------------------------------------------------------

#define B_    8
#define R_    10
#define NROI  80
#define CS    2048
#define CF    256
#define CT    2304      // CS + CF
#define NS    8
#define NF    32
#define H_    16
#define W_    16
#define HW_   256
#define RD_   1024
#define HD_   512
#define NC_   60
#define P49   49

typedef __attribute__((ext_vector_type(16))) __bf16 v16bf;
typedef __attribute__((ext_vector_type(8)))  __bf16 v8bf;
typedef __attribute__((ext_vector_type(8)))  float  v8f;

__device__ __forceinline__ __bf16 tobf(float x) {
  // round-to-nearest-even f32 -> bf16
  unsigned u = __float_as_uint(x);
  unsigned r = (u + 0x7FFFu + ((u >> 16) & 1u)) >> 16;
  unsigned short h = (unsigned short)r;
  return __builtin_bit_cast(__bf16, h);
}

// ---------------------------- B-matrix loaders -----------------------------
// B element (k, n) of the K x N activation matrix. Per-column state is
// resolved once via col(n); load(col, k) must be branchless.

struct BImg {           // activations laid out (img, C, P); n = img*P + p
  const float* p; int C; int P;
  struct Col { const float* base; int P; };
  __device__ __forceinline__ Col col(int n) const {
    Col c; c.base = p + ((size_t)(n / P) * C) * P + (n % P); c.P = P; return c;
  }
  __device__ __forceinline__ float load(const Col& c, int k) const {
    return c.base[k * c.P];
  }
};

struct BConv3 {         // implicit im2col for 3x3 conv; k = ic*9 + kh*3 + kw
  const float* p; int C, IH, IW, OH, OW, pad;
  struct Col { const float* base; int oh, ow, IH, IW, pad; };
  __device__ __forceinline__ Col col(int n) const {
    Col c;
    int img = n / (OH * OW); int r = n % (OH * OW);
    c.base = p + (size_t)img * C * IH * IW;
    c.oh = r / OW; c.ow = r % OW;
    c.IH = IH; c.IW = IW; c.pad = pad;
    return c;
  }
  __device__ __forceinline__ float load(const Col& c, int k) const {
    int ic = k / 9;                  // constant-divisor: magic multiply
    int kr = k - ic * 9;
    int kh = kr / 3;
    int kw = kr - kh * 3;
    int ih = c.oh + kh - c.pad, iw = c.ow + kw - c.pad;
    bool ok = ((unsigned)ih < (unsigned)c.IH) && ((unsigned)iw < (unsigned)c.IW);
    int ihc = min(max(ih, 0), c.IH - 1);
    int iwc = min(max(iw, 0), c.IW - 1);
    float v = c.base[(ic * c.IH + ihc) * c.IW + iwc];  // always in-bounds
    return ok ? v : 0.f;                               // v_cndmask, no branch
  }
};

// ------------------------------- storers -----------------------------------

struct SImg {           // out laid out (img, M, P)
  float* p; int Msz; int P; bool relu;
  __device__ __forceinline__ void store(int m, int n, float v) const {
    if (relu) v = fmaxf(v, 0.f);
    p[((n / P) * Msz + m) * P + (n % P)] = v;
  }
};

struct SImgAdd {        // out += v (residual)
  float* p; int Msz; int P;
  __device__ __forceinline__ void store(int m, int n, float v) const {
    p[((n / P) * Msz + m) * P + (n % P)] += v;
  }
};

struct SNM {            // out[n*ld + off + m], bounded in M
  float* p; int ld; int off; int Mlim; bool relu;
  __device__ __forceinline__ void store(int m, int n, float v) const {
    if (m >= Mlim) return;
    if (relu) v = fmaxf(v, 0.f);
    p[n * ld + off + m] = v;
  }
};

// ------------------------- WMMA GEMM (bf16 -> f32) -------------------------
// C(MxN) = A(MxK, f32 row-major, row stride lda, col offset koff) * B(KxN).
// Block = 256 threads = 8 waves arranged 2(M) x 4(N); each wave owns two
// 16x16 N-tiles -> block tile 32 x 128, K-step 32. A (32x32) and B (32x128)
// tiles staged in LDS as bf16 with an 80-byte row stride: every WMMA
// fragment is two aligned 16B runs (ds_load_b128) and lanes hit distinct
// banks. Each staging thread owns ONE B column for the whole K loop
// (256 = 2 x 128), so div/mod column state is hoisted out of the loop.
// Fragment layouts per CDNA5 ISA 7.12.2 (wave32):
//   A 16x32 bf16 : lane L -> row M = L&15; packed K = two runs of 8 at
//                  kbase = (L>>4)*8 and kbase+16
//   B 32x16 bf16 : lane L -> col N = L&15; packed K = two runs of 8 at
//                  kbase = (L>>4)*8 and kbase+16 (same as A)
//   C/D 16x16 f32: VGPR r -> (M = r + 8*(L>>4), N = L&15)

#define LDSTRIDE 40    // bf16 elements per LDS row (80 bytes)

template <class BL, class ST>
__global__ __launch_bounds__(256)
void wmma_gemm_kernel(const float* __restrict__ A, int lda, int koff,
                      int M, int K, int Ntot, BL bl, ST st) {
  __shared__ alignas(16) __bf16 lsA[32 * LDSTRIDE];
  __shared__ alignas(16) __bf16 lsB[128 * LDSTRIDE];

  const int tid  = threadIdx.x;
  const int wave = tid >> 5;
  const int lane = tid & 31;
  const int ml = lane & 15;
  const int hi = lane >> 4;
  const int wm = wave >> 2;        // 0..1
  const int wn = wave & 3;         // 0..3
  const int mBlk = blockIdx.y * 32;
  const int nBlk = blockIdx.x * 128;

  // A staging: thread -> (row 0..31, 4 consecutive k) ; float4 global load
  const int sa_row  = tid >> 3;
  const int sa_kq   = (tid & 7) << 2;
  const int sa_arow = mBlk + sa_row;
  const float amask = (sa_arow < M) ? 1.f : 0.f;
  const float* Abase = A + (size_t)min(sa_arow, M - 1) * lda + koff;

  // B staging: one column per thread (invariant across K), 8 k-pairs each
  const int nl  = tid & 127;
  const int kpb = tid >> 7;        // 0..1 ; pairs kp = kpb + 2e, e = 0..7
  typename BL::Col bcol = bl.col(min(nBlk + nl, Ntot - 1));
  __bf16* lsBcol = lsB + nl * LDSTRIDE;

  v8f acc0 = {};
  v8f acc1 = {};
  for (int k0 = 0; k0 < K; k0 += 32) {
    // ---- stage A tile (32x32 f32 -> bf16, b128 loads) ----
    {
      float4 w = *(const float4*)(Abase + k0 + sa_kq);
      union { __bf16 h[4]; uint2 u; } pk;
      pk.h[0] = tobf(w.x * amask);
      pk.h[1] = tobf(w.y * amask);
      pk.h[2] = tobf(w.z * amask);
      pk.h[3] = tobf(w.w * amask);
      *(uint2*)(lsA + sa_row * LDSTRIDE + sa_kq) = pk.u;   // 8B aligned
    }
    // ---- stage B tile (column-major in LDS: [n][k]) ----
#pragma unroll
    for (int e = 0; e < 8; ++e) {
      int kp = kpb + 2 * e;          // 0..15
      union { __bf16 h[2]; unsigned u; } pk;
      pk.h[0] = tobf(bl.load(bcol, k0 + 2 * kp));
      pk.h[1] = tobf(bl.load(bcol, k0 + 2 * kp + 1));
      *(unsigned*)(lsBcol + 2 * kp) = pk.u;                // 4B aligned
    }
    __syncthreads();
    // ---- fragments from LDS (two 16B runs each) ----
    const __bf16* ra = lsA + (wm * 16 + ml) * LDSTRIDE + hi * 8;
    v8bf alo = *(const v8bf*)(ra);
    v8bf ahi = *(const v8bf*)(ra + 16);
    v16bf a = __builtin_shufflevector(alo, ahi,
        0, 1, 2, 3, 4, 5, 6, 7, 8, 9, 10, 11, 12, 13, 14, 15);
    const __bf16* rb0 = lsB + (wn * 32 + ml) * LDSTRIDE + hi * 8;
    v8bf b0lo = *(const v8bf*)(rb0);
    v8bf b0hi = *(const v8bf*)(rb0 + 16);
    v16bf b0 = __builtin_shufflevector(b0lo, b0hi,
        0, 1, 2, 3, 4, 5, 6, 7, 8, 9, 10, 11, 12, 13, 14, 15);
    const __bf16* rb1 = lsB + (wn * 32 + 16 + ml) * LDSTRIDE + hi * 8;
    v8bf b1lo = *(const v8bf*)(rb1);
    v8bf b1hi = *(const v8bf*)(rb1 + 16);
    v16bf b1 = __builtin_shufflevector(b1lo, b1hi,
        0, 1, 2, 3, 4, 5, 6, 7, 8, 9, 10, 11, 12, 13, 14, 15);
    acc0 = __builtin_amdgcn_wmma_f32_16x16x32_bf16(
        false, a, false, b0, (short)0, acc0, false, false);
    acc1 = __builtin_amdgcn_wmma_f32_16x16x32_bf16(
        false, a, false, b1, (short)0, acc1, false, false);
    __syncthreads();
  }

  const int m0 = mBlk + wm * 16;
  {
    int n0 = nBlk + wn * 32;
    if (n0 < Ntot) {                 // N tiles are 16-aligned at all call sites
#pragma unroll
      for (int r = 0; r < 8; ++r) {
        int m = m0 + r + hi * 8;
        if (m < M) st.store(m, n0 + ml, acc0[r]);
      }
    }
  }
  {
    int n0 = nBlk + wn * 32 + 16;
    if (n0 < Ntot) {
#pragma unroll
      for (int r = 0; r < 8; ++r) {
        int m = m0 + r + hi * 8;
        if (m < M) st.store(m, n0 + ml, acc1[r]);
      }
    }
  }
}

// ------------------------- ROI align (fused tmean) -------------------------

__device__ __forceinline__ float bilin16(const float* __restrict__ f,
                                         float ys, float xs) {
  bool valid = (ys > -1.f) && (ys < 16.f) && (xs > -1.f) && (xs < 16.f);
  float y = fminf(fmaxf(ys, 0.f), 15.f);
  float x = fminf(fmaxf(xs, 0.f), 15.f);
  int y0 = (int)floorf(y), x0 = (int)floorf(x);
  int y1 = min(y0 + 1, 15), x1 = min(x0 + 1, 15);
  float ly = y - (float)y0, lx = x - (float)x0;
  float hy = 1.f - ly, hx = 1.f - lx;
  float v = f[y0 * 16 + x0] * hy * hx + f[y0 * 16 + x1] * hy * lx +
            f[y1 * 16 + x0] * ly * hx + f[y1 * 16 + x1] * ly * lx;
  return valid ? v : 0.f;
}

__device__ __forceinline__ float roi_point(const float* __restrict__ plane,
                                           float x1, float y1, float x2, float y2,
                                           int py, int px) {
  float rw = fmaxf(x2 - x1, 1.f), rh = fmaxf(y2 - y1, 1.f);
  float bw = rw * (1.f / 7.f), bh = rh * (1.f / 7.f);
  float gw = ceilf(bw), gh = ceilf(bh);
  float s = 0.f;
#pragma unroll
  for (int iy = 0; iy < 3; ++iy) {     // MAXG = ceil(16/7) = 3
    if ((float)iy >= gh) continue;
    float yy = y1 + (float)py * bh + ((float)iy + 0.5f) * bh / gh;
#pragma unroll
    for (int ix = 0; ix < 3; ++ix) {
      if ((float)ix >= gw) continue;
      float xx = x1 + (float)px * bw + ((float)ix + 0.5f) * bw / gw;
      s += bilin16(plane, yy, xx);
    }
  }
  return s / (gh * gw);
}

// pre[n, c, p] : temporal-masked-mean of per-frame ROI-align samples
__global__ void roi_pre_kernel(const float* __restrict__ fs,
                               const float* __restrict__ ff,
                               const float* __restrict__ rois,
                               float* __restrict__ pre) {
  int idx = blockIdx.x * blockDim.x + threadIdx.x;
  const int TOT = NROI * CT * P49;
  if (idx >= TOT) return;
  int n = idx / (CT * P49);
  int r = idx % (CT * P49);
  int c = r / P49;
  int p = r % P49;
  int py = p / 7, px = p % 7;
  bool slow = c < CS;
  int T = slow ? NS : NF;
  float acc = 0.f; int cnt = 0;
  for (int t = 0; t < T; ++t) {
    int troi = slow ? (4 * t + 3) : t;
    const float* rr = rois + ((size_t)n * NF + troi) * 5;
    float x1 = rr[1], y1 = rr[2], x2 = rr[3], y2 = rr[4];
    if (x1 == 1.f && y1 == 1.f && x2 == 1.f && y2 == 1.f) continue;  // nz mask
    ++cnt;
    int b = (int)rr[0];
    const float* plane = slow
        ? fs + ((((size_t)b * CS + c) * NS + t) * HW_)
        : ff + ((((size_t)b * CF + (c - CS)) * NF + t) * HW_);
    acc += roi_point(plane, x1 * 16.f, y1 * 16.f, x2 * 16.f, y2 * 16.f, py, px);
  }
  pre[idx] = acc / fmaxf((float)cnt, 1.f);
}

// gmean[b, c, p] : temporal mean of the global features
__global__ void gmean_kernel(const float* __restrict__ fs,
                             const float* __restrict__ ff,
                             float* __restrict__ g) {
  int idx = blockIdx.x * blockDim.x + threadIdx.x;
  const int TOT = B_ * CT * HW_;
  if (idx >= TOT) return;
  int b = idx / (CT * HW_);
  int r = idx % (CT * HW_);
  int c = r / HW_;
  int p = r % HW_;
  float s = 0.f;
  if (c < CS) {
    const float* base = fs + (((size_t)b * CS + c) * NS) * HW_ + p;
    for (int t = 0; t < NS; ++t) s += base[t * HW_];
    s *= (1.f / NS);
  } else {
    const float* base = ff + (((size_t)b * CF + (c - CS)) * NF) * HW_ + p;
    for (int t = 0; t < NF; ++t) s += base[t * HW_];
    s *= (1.f / NF);
  }
  g[idx] = s;
}

// roi_feats[n, oc] = max over 49 spatial bins
__global__ void max_spatial_kernel(const float* __restrict__ r,
                                   float* __restrict__ o) {
  int idx = blockIdx.x * blockDim.x + threadIdx.x;
  if (idx >= NROI * RD_) return;
  const float* s = r + (size_t)idx * P49;
  float m = s[0];
  for (int p = 1; p < P49; ++p) m = fmaxf(m, s[p]);
  o[idx] = m;
}

// x1[n, oc, p] = relu(gbg[b, oc, p] + abias[n, oc])   (fused conv1 halves)
__global__ void combine_relu_kernel(const float* __restrict__ gbg,
                                    const float* __restrict__ abias,
                                    float* __restrict__ x1) {
  int idx = blockIdx.x * blockDim.x + threadIdx.x;
  const int TOT = NROI * HD_ * HW_;
  if (idx >= TOT) return;
  int n = idx / (HD_ * HW_);
  int r = idx % (HD_ * HW_);
  int oc = r / HW_;
  int p = r % HW_;
  int b = n / R_;
  float v = gbg[((size_t)b * HD_ + oc) * HW_ + p] + abias[(size_t)n * HD_ + oc];
  x1[idx] = fmaxf(v, 0.f);
}

// 3x3 / stride-2 / pad-1 max pool: (80,512,14,14) -> (80,512,7,7)
__global__ void maxpool_kernel(const float* __restrict__ y,
                               float* __restrict__ xo) {
  int idx = blockIdx.x * blockDim.x + threadIdx.x;
  const int TOT = NROI * HD_ * P49;
  if (idx >= TOT) return;
  int p = idx % P49;
  int nc = idx / P49;
  int oh = p / 7, ow = p % 7;
  float m = -3.4e38f;
  for (int dh = 0; dh < 3; ++dh) {
    int ih = oh * 2 - 1 + dh;
    if (ih < 0 || ih >= 14) continue;
    for (int dw = 0; dw < 3; ++dw) {
      int iw = ow * 2 - 1 + dw;
      if (iw < 0 || iw >= 14) continue;
      m = fmaxf(m, y[(size_t)nc * 196 + ih * 14 + iw]);
    }
  }
  xo[idx] = m;
}

// 10x10 channel attention + softmax; one wave32 per (b, i, hw)
__global__ __launch_bounds__(256)
void attn_kernel(const float* __restrict__ q, const float* __restrict__ k,
                 const float* __restrict__ v, float* __restrict__ virt) {
  int wid = (blockIdx.x * blockDim.x + threadIdx.x) >> 5;
  int lane = threadIdx.x & 31;
  const int TOT = B_ * R_ * P49;
  if (wid >= TOT) return;
  int b = wid / (R_ * P49);
  int rr = wid % (R_ * P49);
  int i = rr / P49;
  int hw = rr % P49;
  float qreg[16];
#pragma unroll
  for (int u = 0; u < 16; ++u) {
    int c = lane + u * 32;
    qreg[u] = q[(((size_t)(b * R_ + i) * HD_) + c) * P49 + hw];
  }
  float att[R_];
  const float scale = 0.04419417382415922f;   // 1/sqrt(512)
  for (int j = 0; j < R_; ++j) {
    float s = 0.f;
#pragma unroll
    for (int u = 0; u < 16; ++u) {
      int c = lane + u * 32;
      s += qreg[u] * k[(((size_t)(b * R_ + j) * HD_) + c) * P49 + hw];
    }
#pragma unroll
    for (int off = 16; off > 0; off >>= 1) s += __shfl_xor(s, off, 32);
    att[j] = s * scale;
  }
  float mx = att[0];
  for (int j = 1; j < R_; ++j) mx = fmaxf(mx, att[j]);
  float den = 0.f;
  for (int j = 0; j < R_; ++j) { att[j] = __expf(att[j] - mx); den += att[j]; }
  float inv = 1.f / den;
#pragma unroll
  for (int u = 0; u < 16; ++u) {
    int c = lane + u * 32;
    float s = 0.f;
    for (int j = 0; j < R_; ++j)
      s += att[j] * v[(((size_t)(b * R_ + j) * HD_) + c) * P49 + hw];
    virt[(((size_t)(b * R_ + i) * HD_) + c) * P49 + hw] = s * inv;
  }
}

// per-sample layer-norm over (C,H,W) + scale/shift + relu
__global__ __launch_bounds__(256)
void norm_relu_kernel(const float* __restrict__ virt,
                      const float* __restrict__ g, const float* __restrict__ bb,
                      float* __restrict__ outp) {
  int n = blockIdx.x;
  const int NE = HD_ * P49;
  const float* src = virt + (size_t)n * NE;
  __shared__ float s1[256];
  __shared__ float s2[256];
  float a = 0.f, q = 0.f;
  for (int e = threadIdx.x; e < NE; e += 256) {
    float x = src[e];
    a += x; q += x * x;
  }
  s1[threadIdx.x] = a; s2[threadIdx.x] = q;
  __syncthreads();
  for (int o = 128; o > 0; o >>= 1) {
    if ((int)threadIdx.x < o) {
      s1[threadIdx.x] += s1[threadIdx.x + o];
      s2[threadIdx.x] += s2[threadIdx.x + o];
    }
    __syncthreads();
  }
  float mu = s1[0] * (1.f / NE);
  float var = s2[0] * (1.f / NE) - mu * mu;
  float rs = rsqrtf(var + 1e-5f);
  float* dst = outp + (size_t)n * NE;
  for (int e = threadIdx.x; e < NE; e += 256) {
    int c = e / P49;
    float x = (src[e] - mu) * rs * g[c] + bb[c];
    dst[e] = fmaxf(x, 0.f);
  }
}

// high_order: spatial mean -> second half of cat buffer
__global__ void high_kernel(const float* __restrict__ x,
                            float* __restrict__ cat) {
  int idx = blockIdx.x * blockDim.x + threadIdx.x;
  if (idx >= NROI * HD_) return;
  int n = idx / HD_, c = idx % HD_;
  const float* s = x + ((size_t)n * HD_ + c) * P49;
  float a = 0.f;
  for (int p = 0; p < P49; ++p) a += s[p];
  cat[(size_t)n * (2 * HD_) + HD_ + c] = a * (1.f / P49);
}

// ---------------------------------------------------------------------------

extern "C" void kernel_launch(void* const* d_in, const int* in_sizes, int n_in,
                              void* d_out, int out_size, void* d_ws, size_t ws_size,
                              hipStream_t stream) {
  const float* feat_slow = (const float*)d_in[0];
  const float* feat_fast = (const float*)d_in[1];
  const float* rois      = (const float*)d_in[2];
  const float* w_reduce  = (const float*)d_in[6];   // (1024, 2304)
  const float* w_conv1   = (const float*)d_in[7];   // (512, 2048)
  const float* w_conv2   = (const float*)d_in[8];   // (512, 4608)
  const float* wq        = (const float*)d_in[9];   // (2, 512, 4608)
  const float* wk        = (const float*)d_in[10];
  const float* wv        = (const float*)d_in[11];
  const float* wm        = (const float*)d_in[12];
  const float* gn_g      = (const float*)d_in[13];  // (2, 512)
  const float* gn_b      = (const float*)d_in[14];
  const float* fc1_w     = (const float*)d_in[15];  // (512, 1024)
  const float* fc2_w     = (const float*)d_in[16];  // (60, 1024)
  float* out = (float*)d_out;

  // ---- workspace carve (floats) ----
  float* W = (float*)d_ws;
  size_t o = 0;
  auto carve = [&](size_t nfl) { float* p = W + o; o += nfl; return p; };
  float* pre     = carve((size_t)NROI * CT * P49);    // 9,031,680
  float* gmean   = carve((size_t)B_ * CT * HW_);      // 4,718,592
  float* red_out = carve((size_t)NROI * RD_ * P49);   // 4,014,080
  float* roifeat = carve((size_t)NROI * RD_);         // 81,920
  float* gfeat   = carve((size_t)B_ * RD_ * HW_);     // 2,097,152
  float* gbg     = carve((size_t)B_ * HD_ * HW_);     // 1,048,576
  float* abias   = carve((size_t)NROI * HD_);         // 40,960
  float* x1      = carve((size_t)NROI * HD_ * HW_);   // 10,485,760
  float* y2      = carve((size_t)NROI * HD_ * 196);   // 8,028,160
  float* x       = carve((size_t)NROI * HD_ * P49);   // 2,007,040
  float* qb      = carve((size_t)NROI * HD_ * P49);
  float* kb      = carve((size_t)NROI * HD_ * P49);
  float* vb      = carve((size_t)NROI * HD_ * P49);
  float* virt    = carve((size_t)NROI * HD_ * P49);
  float* vrn     = carve((size_t)NROI * HD_ * P49);
  float* cat     = carve((size_t)NROI * 2 * HD_);     // 81,920
  (void)in_sizes; (void)n_in; (void)out_size; (void)ws_size;

  const dim3 blk(256);

  // 1) ROI-align + temporal masked mean -> pre (80, 2304, 49)
  {
    int tot = NROI * CT * P49;
    roi_pre_kernel<<<(tot + 255) / 256, blk, 0, stream>>>(feat_slow, feat_fast, rois, pre);
  }
  // 2) temporal mean of global features -> gmean (8, 2304, 256)
  {
    int tot = B_ * CT * HW_;
    gmean_kernel<<<(tot + 255) / 256, blk, 0, stream>>>(feat_slow, feat_fast, gmean);
  }
  // 3) w_reduce GEMM on ROI features: M=1024, K=2304, N=3920
  {
    BImg bl{pre, CT, P49};
    SImg st{red_out, RD_, P49, false};
    dim3 g((NROI * P49 + 127) / 128, (RD_ + 31) / 32);
    wmma_gemm_kernel<<<g, blk, 0, stream>>>(w_reduce, CT, 0, RD_, CT, NROI * P49, bl, st);
  }
  // 4) spatial max -> roifeat (80, 1024)
  max_spatial_kernel<<<(NROI * RD_ + 255) / 256, blk, 0, stream>>>(red_out, roifeat);
  // 5) w_reduce GEMM on global features: M=1024, K=2304, N=2048 -> gfeat
  {
    BImg bl{gmean, CT, HW_};
    SImg st{gfeat, RD_, HW_, false};
    dim3 g((B_ * HW_ + 127) / 128, (RD_ + 31) / 32);
    wmma_gemm_kernel<<<g, blk, 0, stream>>>(w_reduce, CT, 0, RD_, CT, B_ * HW_, bl, st);
  }
  // 6) conv1 left half (context): M=512, K=1024, N=2048 -> gbg
  {
    BImg bl{gfeat, RD_, HW_};
    SImg st{gbg, HD_, HW_, false};
    dim3 g((B_ * HW_ + 127) / 128, (HD_ + 31) / 32);
    wmma_gemm_kernel<<<g, blk, 0, stream>>>(w_conv1, 2 * RD_, 0, HD_, RD_, B_ * HW_, bl, st);
  }
  // 7) conv1 right half (actor, spatially constant): M=512, K=1024, N=80 -> abias
  {
    BImg bl{roifeat, RD_, 1};
    SNM st{abias, HD_, 0, HD_, false};
    dim3 g((NROI + 127) / 128, (HD_ + 31) / 32);
    wmma_gemm_kernel<<<g, blk, 0, stream>>>(w_conv1, 2 * RD_, RD_, HD_, RD_, NROI, bl, st);
  }
  // 8) combine + relu -> x1 (80, 512, 16, 16)
  {
    int tot = NROI * HD_ * HW_;
    combine_relu_kernel<<<(tot + 255) / 256, blk, 0, stream>>>(gbg, abias, x1);
  }
  // 9) conv2 3x3 pad0 implicit GEMM: M=512, K=4608, N=15680 -> relu -> y2
  {
    BConv3 bl{x1, HD_, 16, 16, 14, 14, 0};
    SImg st{y2, HD_, 196, true};
    dim3 g((NROI * 196 + 127) / 128, (HD_ + 31) / 32);
    wmma_gemm_kernel<<<g, blk, 0, stream>>>(w_conv2, HD_ * 9, 0, HD_, HD_ * 9, NROI * 196, bl, st);
  }
  // 10) maxpool 3x3 s2 p1 -> x (80, 512, 7, 7)
  {
    int tot = NROI * HD_ * P49;
    maxpool_kernel<<<(tot + 255) / 256, blk, 0, stream>>>(y2, x);
  }
  // 11) HR2O rounds
  for (int d = 0; d < 2; ++d) {
    size_t woff = (size_t)d * HD_ * HD_ * 9;
    BConv3 bl{x, HD_, 7, 7, 7, 7, 1};
    dim3 g((NROI * P49 + 127) / 128, (HD_ + 31) / 32);
    {
      SImg st{qb, HD_, P49, false};
      wmma_gemm_kernel<<<g, blk, 0, stream>>>(wq + woff, HD_ * 9, 0, HD_, HD_ * 9, NROI * P49, bl, st);
    }
    {
      SImg st{kb, HD_, P49, false};
      wmma_gemm_kernel<<<g, blk, 0, stream>>>(wk + woff, HD_ * 9, 0, HD_, HD_ * 9, NROI * P49, bl, st);
    }
    {
      SImg st{vb, HD_, P49, false};
      wmma_gemm_kernel<<<g, blk, 0, stream>>>(wv + woff, HD_ * 9, 0, HD_, HD_ * 9, NROI * P49, bl, st);
    }
    // attention: one wave per (b, i, hw) -> 3920 waves
    attn_kernel<<<(B_ * R_ * P49 + 7) / 8, blk, 0, stream>>>(qb, kb, vb, virt);
    // per-sample norm + affine + relu
    norm_relu_kernel<<<NROI, blk, 0, stream>>>(virt, gn_g + d * HD_, gn_b + d * HD_, vrn);
    // wm conv + residual add into x
    {
      BConv3 blm{vrn, HD_, 7, 7, 7, 7, 1};
      SImgAdd st{x, HD_, P49};
      wmma_gemm_kernel<<<g, blk, 0, stream>>>(wm + woff, HD_ * 9, 0, HD_, HD_ * 9, NROI * P49, blm, st);
    }
  }
  // 12) high-order spatial mean -> cat[:, 512:1024]
  high_kernel<<<(NROI * HD_ + 255) / 256, blk, 0, stream>>>(x, cat);
  // 13) fc1 + relu -> cat[:, 0:512]  (M=512, K=1024, N=80)
  {
    BImg bl{roifeat, RD_, 1};
    SNM st{cat, 2 * HD_, 0, HD_, true};
    dim3 g((NROI + 127) / 128, (HD_ + 31) / 32);
    wmma_gemm_kernel<<<g, blk, 0, stream>>>(fc1_w, RD_, 0, HD_, RD_, NROI, bl, st);
  }
  // 14) fc2 -> out (80, 60)  (M=60 padded to 64 via guards, K=1024, N=80)
  {
    BImg bl{cat, 2 * HD_, 1};
    SNM st{out, NC_, 0, NC_, false};
    dim3 g((NROI + 127) / 128, (NC_ + 31) / 32);
    wmma_gemm_kernel<<<g, blk, 0, stream>>>(fc2_w, 2 * HD_, 0, NC_, 2 * HD_, NROI, bl, st);
  }
}
